// ExpertFFN_71528385347979
// MI455X (gfx1250) — matrix operand order repaired and checked by reading, not compiled, not run on hardware
//
#include <hip/hip_runtime.h>
#include <hip/hip_bf16.h>

#define HIDDEN 4096
#define INTER  11008
#define TOKENS 8192
#define LDA    40      // padded LDS row stride in bf16 (80B) -> conflict-free b128 frag loads

// knobs (flip if a probe fails to compile)
#ifndef USE_CVT_ASM
#define USE_CVT_ASM 1            // probe: native v_cvt_pk_bf16_f32 via inline asm
#endif
#ifndef USE_ASYNC_LDS
#define USE_ASYNC_LDS 1
#endif

typedef __attribute__((ext_vector_type(16))) __bf16        v16bf;
typedef __attribute__((ext_vector_type(2)))  __bf16        v2bf;
typedef __attribute__((ext_vector_type(8)))  float         v8f;
typedef __attribute__((ext_vector_type(4)))  unsigned int  v4u;
typedef __attribute__((ext_vector_type(8)))  unsigned int  v8u;

static_assert(TOKENS % 128 == 0 && INTER % 64 == 0 && HIDDEN % 128 == 0, "tiling");
static_assert(HIDDEN % 32 == 0 && INTER % 32 == 0, "k tiling");

// fp32 -> bf16 RNE, two at a time, packed into one dword
__device__ __forceinline__ unsigned int cvt2_bf16(float a, float b) {
#if __has_builtin(__builtin_amdgcn_cvt_pk_bf16_f32)
  v2bf r = __builtin_amdgcn_cvt_pk_bf16_f32(a, b);
  return __builtin_bit_cast(unsigned int, r);
#elif USE_CVT_ASM
  unsigned int r;
  asm("v_cvt_pk_bf16_f32 %0, %1, %2" : "=v"(r) : "v"(a), "v"(b));
  return r;
#else
  unsigned int ua = __float_as_uint(a);
  unsigned int ub = __float_as_uint(b);
  ua = ua + 0x7FFFu + ((ua >> 16) & 1u);
  ub = ub + 0x7FFFu + ((ub >> 16) & 1u);
  return __builtin_amdgcn_perm(ub, ua, 0x07060302);  // hi16(ua) | hi16(ub)<<16
#endif
}
__device__ __forceinline__ unsigned short cvt1_bf16(float a) {
#if USE_CVT_ASM
  return (unsigned short)(cvt2_bf16(a, a) & 0xFFFFu);
#else
  unsigned int ua = __float_as_uint(a);
  return (unsigned short)((ua + 0x7FFFu + ((ua >> 16) & 1u)) >> 16);
#endif
}

// Assemble an 8-VGPR WMMA operand fragment from two 16B LDS chunks.
__device__ __forceinline__ v16bf frag16(const __bf16* p0, const __bf16* p1) {
  v4u lo = *(const v4u*)p0;
  v4u hi = *(const v4u*)p1;
  v8u v  = __builtin_shufflevector(lo, hi, 0, 1, 2, 3, 4, 5, 6, 7);
  return __builtin_bit_cast(v16bf, v);
}

// ---------------------------------------------------------------------------
// Kernel 1: gate = x @ Wg^T ; up = x @ Wu^T ; hidden = silu(gate)*up  (bf16 out)
// block: 256 thr (8 waves). Block tile 128(M) x 64(N). Wave tile 32x32 (x2 mats).
// ---------------------------------------------------------------------------
__global__ __launch_bounds__(256)
void ffn_gate_up_kernel(const float* __restrict__ x,
                        const float* __restrict__ Wg,
                        const float* __restrict__ Wu,
                        unsigned short* __restrict__ hidden) {
  __shared__ __align__(16) __bf16 sA[2][128 * LDA];
  __shared__ __align__(16) __bf16 sG[2][64 * LDA];
  __shared__ __align__(16) __bf16 sU[2][64 * LDA];

  const int tid   = threadIdx.x;
  const int lane  = tid & 31;
  const int wave  = tid >> 5;
  const int waveM = wave >> 1;          // 0..3  -> 32-row slab
  const int waveN = wave & 1;           // 0..1  -> 32-col slab
  const int m0 = blockIdx.y * 128;
  const int n0 = blockIdx.x * 64;

  // producer mapping: A tile 128x32 f32 (16 f32/thr), B tiles 64x32 f32 (8 f32/thr each)
  const int arow = tid >> 1, aseg = tid & 1;   // aseg*16 floats
  const int brow = tid >> 2, bseg = tid & 3;   // bseg*8 floats

  const float* gpa = x  + (size_t)(m0 + arow) * HIDDEN + aseg * 16;
  const float* gpg = Wg + (size_t)(n0 + brow) * HIDDEN + bseg * 8;
  const float* gpu = Wu + (size_t)(n0 + brow) * HIDDEN + bseg * 8;

  float4 ra[4], rg[2], ru[2];

  auto load_tile = [&](int kt) {
    const float* pa = gpa + kt * 32;
    ra[0] = ((const float4*)pa)[0]; ra[1] = ((const float4*)pa)[1];
    ra[2] = ((const float4*)pa)[2]; ra[3] = ((const float4*)pa)[3];
    const float* pg = gpg + kt * 32;
    rg[0] = ((const float4*)pg)[0]; rg[1] = ((const float4*)pg)[1];
    const float* pu = gpu + kt * 32;
    ru[0] = ((const float4*)pu)[0]; ru[1] = ((const float4*)pu)[1];
  };

  auto store_tile = [&](int buf) {
    v4u w0, w1;
    w0.x = cvt2_bf16(ra[0].x, ra[0].y); w0.y = cvt2_bf16(ra[0].z, ra[0].w);
    w0.z = cvt2_bf16(ra[1].x, ra[1].y); w0.w = cvt2_bf16(ra[1].z, ra[1].w);
    w1.x = cvt2_bf16(ra[2].x, ra[2].y); w1.y = cvt2_bf16(ra[2].z, ra[2].w);
    w1.z = cvt2_bf16(ra[3].x, ra[3].y); w1.w = cvt2_bf16(ra[3].z, ra[3].w);
    __bf16* da = &sA[buf][arow * LDA + aseg * 16];
    ((v4u*)da)[0] = w0; ((v4u*)da)[1] = w1;

    v4u g;
    g.x = cvt2_bf16(rg[0].x, rg[0].y); g.y = cvt2_bf16(rg[0].z, rg[0].w);
    g.z = cvt2_bf16(rg[1].x, rg[1].y); g.w = cvt2_bf16(rg[1].z, rg[1].w);
    *(v4u*)&sG[buf][brow * LDA + bseg * 8] = g;
    v4u u;
    u.x = cvt2_bf16(ru[0].x, ru[0].y); u.y = cvt2_bf16(ru[0].z, ru[0].w);
    u.z = cvt2_bf16(ru[1].x, ru[1].y); u.w = cvt2_bf16(ru[1].z, ru[1].w);
    *(v4u*)&sU[buf][brow * LDA + bseg * 8] = u;
  };

  v8f cg[2][2] = {}; v8f cu[2][2] = {};

  const int l16   = lane & 15;
  const int koffA = (lane >> 4) * 16;  // A: 16 contiguous K at koffA
  const int koffB = (lane >> 4) * 16;  // B: 16 contiguous K at koffB

  load_tile(0);
  store_tile(0);
  __syncthreads();

  const int KT = HIDDEN / 32;   // 128
  for (int kt = 0; kt < KT; ++kt) {
    const int cur = kt & 1;
    if (kt + 1 < KT) load_tile(kt + 1);

    v16bf af[2], gf[2], uf[2];
#pragma unroll
    for (int mf = 0; mf < 2; ++mf) {
      const __bf16* base = &sA[cur][(waveM * 32 + mf * 16 + l16) * LDA];
      af[mf] = frag16(base + koffA, base + koffA + 8);
    }
#pragma unroll
    for (int nf = 0; nf < 2; ++nf) {
      const __bf16* bg = &sG[cur][(waveN * 32 + nf * 16 + l16) * LDA];
      gf[nf] = frag16(bg + koffB, bg + koffB + 8);
      const __bf16* bu = &sU[cur][(waveN * 32 + nf * 16 + l16) * LDA];
      uf[nf] = frag16(bu + koffB, bu + koffB + 8);
    }
#pragma unroll
    for (int mf = 0; mf < 2; ++mf)
#pragma unroll
      for (int nf = 0; nf < 2; ++nf) {
        cg[mf][nf] = __builtin_amdgcn_wmma_f32_16x16x32_bf16(
            false, af[mf], false, gf[nf], (short)0, cg[mf][nf], false, false);
        cu[mf][nf] = __builtin_amdgcn_wmma_f32_16x16x32_bf16(
            false, af[mf], false, uf[nf], (short)0, cu[mf][nf], false, false);
      }

    if (kt + 1 < KT) store_tile(cur ^ 1);
    __syncthreads();
  }

  // epilogue: silu(gate)*up -> bf16 hidden. C frag: lane l, vgpr r -> (M=r+8*(l>>4), N=l&15)
  const int crow = (lane >> 4) * 8;
  const int ccol = lane & 15;
#pragma unroll
  for (int mf = 0; mf < 2; ++mf)
#pragma unroll
    for (int nf = 0; nf < 2; ++nf) {
      const int gr = m0 + waveM * 32 + mf * 16 + crow;
      const int gc = n0 + waveN * 32 + nf * 16 + ccol;
#pragma unroll
      for (int r = 0; r < 8; ++r) {
        float g = cg[mf][nf][r];
        float u = cu[mf][nf][r];
        float h = g * u * __builtin_amdgcn_rcpf(1.0f + __expf(-g));  // SiLU(g)*u
        hidden[(size_t)(gr + r) * INTER + gc] = cvt1_bf16(h);
      }
    }
}

// ---------------------------------------------------------------------------
// Kernel 2: out = hidden @ Wd^T (hidden bf16 via async->LDS, Wd f32 -> bf16)
// block: 256 thr (8 waves). Block tile 128(M) x 128(N). Wave tile 32x64.
// ---------------------------------------------------------------------------
__global__ __launch_bounds__(256)
void ffn_down_kernel(const unsigned short* __restrict__ hidden,
                     const float* __restrict__ Wd,
                     float* __restrict__ out) {
  __shared__ __align__(16) __bf16 sA[2][128 * LDA];
  __shared__ __align__(16) __bf16 sB[2][128 * LDA];

  const int tid   = threadIdx.x;
  const int lane  = tid & 31;
  const int wave  = tid >> 5;
  const int waveM = wave >> 1;          // 0..3
  const int waveN = wave & 1;           // 0..1 -> 64-col slab
  const int m0 = blockIdx.y * 128;
  const int n0 = blockIdx.x * 128;

  const int arow = tid >> 1, aseg = tid & 1;  // both tiles: 128 rows x 32 k, 16 elems/thr

  const unsigned short* gha = hidden + (size_t)(m0 + arow) * INTER + aseg * 16;
  const float*          gwb = Wd + (size_t)(n0 + arow) * INTER + aseg * 16;

#if !USE_ASYNC_LDS
  v4u rha[2];
#endif
  float4 rb[4];       // f32 B tile (16 floats)

  // A tile: bf16 passthrough — async copy global -> LDS (ASYNCcnt), no VGPR staging
  auto copyA_async = [&](int kt, int buf) {
#if USE_ASYNC_LDS
    const unsigned short* src = gha + kt * 32;
    unsigned int dst = (unsigned int)(uintptr_t)&sA[buf][arow * LDA + aseg * 16];
    asm volatile("global_load_async_to_lds_b128 %0, %1, off"
                 :: "v"(dst), "v"(src) : "memory");
    asm volatile("global_load_async_to_lds_b128 %0, %1, off offset:16"
                 :: "v"(dst + 16), "v"(src) : "memory");
#else
    const v4u* pa = (const v4u*)(gha + kt * 32);
    rha[0] = pa[0]; rha[1] = pa[1];
#endif
  };

  auto loadB = [&](int kt) {
    const float* pb = gwb + kt * 32;
    rb[0] = ((const float4*)pb)[0]; rb[1] = ((const float4*)pb)[1];
    rb[2] = ((const float4*)pb)[2]; rb[3] = ((const float4*)pb)[3];
  };

  auto storeB = [&](int buf) {
#if !USE_ASYNC_LDS
    __bf16* da = &sA[buf][arow * LDA + aseg * 16];
    ((v4u*)da)[0] = rha[0]; ((v4u*)da)[1] = rha[1];
#endif
    v4u w0, w1;
    w0.x = cvt2_bf16(rb[0].x, rb[0].y); w0.y = cvt2_bf16(rb[0].z, rb[0].w);
    w0.z = cvt2_bf16(rb[1].x, rb[1].y); w0.w = cvt2_bf16(rb[1].z, rb[1].w);
    w1.x = cvt2_bf16(rb[2].x, rb[2].y); w1.y = cvt2_bf16(rb[2].z, rb[2].w);
    w1.z = cvt2_bf16(rb[3].x, rb[3].y); w1.w = cvt2_bf16(rb[3].z, rb[3].w);
    __bf16* db = &sB[buf][arow * LDA + aseg * 16];
    ((v4u*)db)[0] = w0; ((v4u*)db)[1] = w1;
  };

  v8f c[2][4] = {};

  const int l16   = lane & 15;
  const int koffA = (lane >> 4) * 16;
  const int koffB = (lane >> 4) * 16;

  copyA_async(0, 0);
  loadB(0);
  storeB(0);
#if USE_ASYNC_LDS
  asm volatile("s_wait_asynccnt 0x0" ::: "memory");
#endif
  __syncthreads();

  const int KT = INTER / 32;   // 344
  for (int kt = 0; kt < KT; ++kt) {
    const int cur = kt & 1;
    if (kt + 1 < KT) {
      copyA_async(kt + 1, cur ^ 1);   // writes a buffer retired by the last barrier
      loadB(kt + 1);
    }

    v16bf af[2], bf[4];
#pragma unroll
    for (int mf = 0; mf < 2; ++mf) {
      const __bf16* base = &sA[cur][(waveM * 32 + mf * 16 + l16) * LDA];
      af[mf] = frag16(base + koffA, base + koffA + 8);
    }
#pragma unroll
    for (int nf = 0; nf < 4; ++nf) {
      const __bf16* base = &sB[cur][(waveN * 64 + nf * 16 + l16) * LDA];
      bf[nf] = frag16(base + koffB, base + koffB + 8);
    }
#pragma unroll
    for (int mf = 0; mf < 2; ++mf)
#pragma unroll
      for (int nf = 0; nf < 4; ++nf)
        c[mf][nf] = __builtin_amdgcn_wmma_f32_16x16x32_bf16(
            false, af[mf], false, bf[nf], (short)0, c[mf][nf], false, false);

    if (kt + 1 < KT) storeB(cur ^ 1);
#if USE_ASYNC_LDS
    asm volatile("s_wait_asynccnt 0x0" ::: "memory");
#endif
    __syncthreads();
  }

  const int crow = (lane >> 4) * 8;
  const int ccol = lane & 15;
#pragma unroll
  for (int mf = 0; mf < 2; ++mf)
#pragma unroll
    for (int nf = 0; nf < 4; ++nf) {
      const int gr = m0 + waveM * 32 + mf * 16 + crow;
      const int gc = n0 + waveN * 64 + nf * 16 + ccol;
#pragma unroll
      for (int r = 0; r < 8; ++r)
        out[(size_t)(gr + r) * HIDDEN + gc] = c[mf][nf][r];
    }
}

// ---------------------------------------------------------------------------
extern "C" void kernel_launch(void* const* d_in, const int* in_sizes, int n_in,
                              void* d_out, int out_size, void* d_ws, size_t ws_size,
                              hipStream_t stream) {
  const float* x  = (const float*)d_in[0];
  const float* Wg = (const float*)d_in[1];
  const float* Wu = (const float*)d_in[2];
  const float* Wd = (const float*)d_in[3];
  float* out = (float*)d_out;

  // workspace: hidden activation (tokens x inter) in bf16 = ~180 MB
  unsigned short* hidden = (unsigned short*)d_ws;

  dim3 g1(INTER / 64, TOKENS / 128);    // 172 x 64 blocks
  ffn_gate_up_kernel<<<g1, 256, 0, stream>>>(x, Wg, Wu, hidden);

  dim3 g2(HIDDEN / 128, TOKENS / 128);  // 32 x 64 blocks
  ffn_down_kernel<<<g2, 256, 0, stream>>>(hidden, Wd, out);
}
